// MS_SSIM_49143015800994
// MI455X (gfx1250) — matrix operand order repaired and checked by reading, not compiled, not run on hardware
//
#include <hip/hip_runtime.h>
#include <math.h>

typedef __attribute__((ext_vector_type(16))) _Float16 v16h;
typedef __attribute__((ext_vector_type(8)))  _Float16 vh8;
typedef __attribute__((ext_vector_type(8)))  float    v8f;
typedef __attribute__((ext_vector_type(4)))  float    vf4;
typedef __attribute__((ext_vector_type(2)))  float    vf2;

#define C1_CONST 6.5025f   // (0.01*255)^2
#define C2_CONST 58.5225f  // (0.03*255)^2
#define HPAD 16            // per-row pad of tmpT: data lives at [8, H+8)

// 1D Gaussian, window 11, sigma 1.5, normalized
__constant__ float GW[11] = {
    0.0010284f, 0.0075988f, 0.0360008f, 0.1093607f, 0.2130055f,
    0.2660117f, 0.2130055f, 0.1093607f, 0.0360008f, 0.0075988f, 0.0010284f
};

__device__ __forceinline__ v8f wmma_f32_16x16x32_f16(v16h a, v16h b, v8f c) {
    return __builtin_amdgcn_wmma_f32_16x16x32_f16(
        /*neg_a=*/false, a, /*neg_b=*/false, b,
        /*c_mod=*/(short)0, c, /*reuse_a=*/false, /*reuse_b=*/false);
}

// Shared banded-weight B fragment (32x16 f16): B[m][c] = g[m - c - 3]
// B layout: element h -> K = h + (h>=8 ? 8 : 0) (+8 for hi lane group), column = lane&15.
__device__ __forceinline__ v16h make_weightB(int lg, int ln) {
    v16h bw;
#pragma unroll
    for (int h = 0; h < 16; ++h) {
        int m  = h + (h >= 8 ? 8 : 0) + (lg ? 8 : 0);
        int gi = m - ln - 3;
        bw[h] = (_Float16)((gi >= 0 && gi <= 10) ? GW[gi] : 0.0f);
    }
    return bw;
}

__global__ void msssim_init(float* acc) {
    if (threadIdx.x < 10) acc[threadIdx.x] = 0.0f;
}

// ---------------------------------------------------------------------------
// Horizontal pass. Per 16x16 tile: D = A(16x32 image rows, K window starts at
// tile_col-8) x B(banded gaussian). Channels x1,x2,x1^2,x2^2,x1*x2.
// Output written TRANSPOSED per channel: tmpT[(b*W + col)*Hp + row + 8],
// so each lane emits one b128 store per channel.
// ---------------------------------------------------------------------------
__global__ void msssim_hconv(const float* __restrict__ i1, const float* __restrict__ i2,
                             _Float16* __restrict__ t1,  _Float16* __restrict__ t2,
                             _Float16* __restrict__ t11, _Float16* __restrict__ t22,
                             _Float16* __restrict__ t12,
                             int H, int W, int nTiles)
{
    int wave = (int)((blockIdx.x * blockDim.x + threadIdx.x) >> 5);
    if (wave >= nTiles) return;                 // wave-uniform guard
    int lane = threadIdx.x & 31;
    int lg   = lane >> 4;
    int ln   = lane & 15;
    int k0   = lg ? 8 : 0;

    int tilesX = W >> 4;
    int tilesY = H >> 4;
    int b  = wave / (tilesX * tilesY);
    int rr = wave % (tilesX * tilesY);
    int ty = rr / tilesX;
    int tx = rr % tilesX;

    size_t imgBase = (size_t)b * (size_t)H * (size_t)W;
    const float* r1 = i1 + imgBase + (size_t)(ty * 16 + ln) * W + (tx * 16 - 8);
    const float* r2 = i2 + imgBase + (size_t)(ty * 16 + ln) * W + (tx * 16 - 8);

    float x1v[16], x2v[16];
    bool interior = (tx > 0) && (tx < tilesX - 1);
    if (interior) {
        // cols tx*16-8+k0 ... all in range; 16B-aligned float4 chunks
        vf4 q0 = *(const vf4*)(r1 + k0);
        vf4 q1 = *(const vf4*)(r1 + k0 + 4);
        vf4 q2 = *(const vf4*)(r1 + k0 + 16);
        vf4 q3 = *(const vf4*)(r1 + k0 + 20);
        vf4 p0 = *(const vf4*)(r2 + k0);
        vf4 p1 = *(const vf4*)(r2 + k0 + 4);
        vf4 p2 = *(const vf4*)(r2 + k0 + 16);
        vf4 p3 = *(const vf4*)(r2 + k0 + 20);
#pragma unroll
        for (int j = 0; j < 4; ++j) {
            x1v[j]      = q0[j]; x1v[4 + j]  = q1[j];
            x1v[8 + j]  = q2[j]; x1v[12 + j] = q3[j];
            x2v[j]      = p0[j]; x2v[4 + j]  = p1[j];
            x2v[8 + j]  = p2[j]; x2v[12 + j] = p3[j];
        }
    } else {
#pragma unroll
        for (int h = 0; h < 16; ++h) {
            int K   = h + (h >= 8 ? 8 : 0) + k0;     // A-layout K mapping
            int col = tx * 16 - 8 + K;
            float x1 = 0.0f, x2 = 0.0f;
            if (col >= 0 && col < W) { x1 = r1[K]; x2 = r2[K]; }
            x1v[h] = x1; x2v[h] = x2;
        }
    }

    v16h a1, a2, a11, a22, a12;
#pragma unroll
    for (int h = 0; h < 16; ++h) {
        float x1 = x1v[h], x2 = x2v[h];
        a1[h]  = (_Float16)x1;
        a2[h]  = (_Float16)x2;
        a11[h] = (_Float16)(x1 * x1);
        a22[h] = (_Float16)(x2 * x2);
        a12[h] = (_Float16)(x1 * x2);
    }
    v16h bw = make_weightB(lg, ln);

    v8f z = {0.f, 0.f, 0.f, 0.f, 0.f, 0.f, 0.f, 0.f};
    v8f d1  = wmma_f32_16x16x32_f16(a1,  bw, z);
    v8f d2  = wmma_f32_16x16x32_f16(a2,  bw, z);
    v8f d11 = wmma_f32_16x16x32_f16(a11, bw, z);
    v8f d22 = wmma_f32_16x16x32_f16(a22, bw, z);
    v8f d12 = wmma_f32_16x16x32_f16(a12, bw, z);

    // Lane owns column (tx*16+ln), rows ty*16 + lg*8 + [0..8): contiguous in tmpT
    int Hp = H + HPAD;
    size_t sidx = (size_t)(b * W + tx * 16 + ln) * (size_t)Hp + (size_t)(ty * 16 + lg * 8 + 8);
    vh8 o1, o2, o11, o22, o12;
#pragma unroll
    for (int i = 0; i < 8; ++i) {
        o1[i]  = (_Float16)d1[i];
        o2[i]  = (_Float16)d2[i];
        o11[i] = (_Float16)d11[i];
        o22[i] = (_Float16)d22[i];
        o12[i] = (_Float16)d12[i];
    }
    *(vh8*)(t1  + sidx) = o1;
    *(vh8*)(t2  + sidx) = o2;
    *(vh8*)(t11 + sidx) = o11;
    *(vh8*)(t22 + sidx) = o22;
    *(vh8*)(t12 + sidx) = o12;
}

// ---------------------------------------------------------------------------
// Vertical pass + SSIM map + reduction. Conv along original rows == conv along
// contiguous axis of tmpT. A = tmpT rows (16x32, two aligned b128 chunks per
// lane per channel), B = banded gaussian. D is the (transposed) tile; only its
// mean is needed.
// ---------------------------------------------------------------------------
__global__ void msssim_vconv(const _Float16* __restrict__ t1,  const _Float16* __restrict__ t2,
                             const _Float16* __restrict__ t11, const _Float16* __restrict__ t22,
                             const _Float16* __restrict__ t12,
                             float* __restrict__ accS, float* __restrict__ accC,
                             int H, int W, int nTiles)
{
    int wave = (int)((blockIdx.x * blockDim.x + threadIdx.x) >> 5);
    if (wave >= nTiles) return;                 // wave-uniform guard
    int lane = threadIdx.x & 31;
    int lg   = lane >> 4;
    int ln   = lane & 15;
    int k0   = lg ? 8 : 0;

    int tilesX = W >> 4;
    int tilesY = H >> 4;
    int b  = wave / (tilesX * tilesY);
    int rr = wave % (tilesX * tilesY);
    int ty = rr / tilesX;
    int tx = rr % tilesX;
    int r0 = ty * 16;
    int Hp = H + HPAD;

    size_t rowBase = (size_t)(b * W + tx * 16 + ln) * (size_t)Hp;
    const _Float16* p1  = t1  + rowBase;
    const _Float16* p2  = t2  + rowBase;
    const _Float16* p11 = t11 + rowBase;
    const _Float16* p22 = t22 + rowBase;
    const _Float16* p12 = t12 + rowBase;

    v16h b1, b2, b11, b22, b12;
    bool interior = (ty > 0) && (ty < tilesY - 1);
    if (interior) {
        // aligned b128 chunks at tmpT cols r0+k0 and r0+k0+16
        vh8 l1 = *(const vh8*)(p1  + r0 + k0), h1 = *(const vh8*)(p1  + r0 + k0 + 16);
        vh8 l2 = *(const vh8*)(p2  + r0 + k0), h2 = *(const vh8*)(p2  + r0 + k0 + 16);
        vh8 l3 = *(const vh8*)(p11 + r0 + k0), h3 = *(const vh8*)(p11 + r0 + k0 + 16);
        vh8 l4 = *(const vh8*)(p22 + r0 + k0), h4 = *(const vh8*)(p22 + r0 + k0 + 16);
        vh8 l5 = *(const vh8*)(p12 + r0 + k0), h5 = *(const vh8*)(p12 + r0 + k0 + 16);
#pragma unroll
        for (int j = 0; j < 8; ++j) {
            b1[j] = l1[j];  b1[8 + j] = h1[j];
            b2[j] = l2[j];  b2[8 + j] = h2[j];
            b11[j] = l3[j]; b11[8 + j] = h3[j];
            b22[j] = l4[j]; b22[8 + j] = h4[j];
            b12[j] = l5[j]; b12[8 + j] = h5[j];
        }
    } else {
#pragma unroll
        for (int h = 0; h < 16; ++h) {
            int K   = h + (h >= 8 ? 8 : 0) + k0;     // A-layout K mapping
            int idx = r0 + K;                        // tmpT col; data at [8, H+8)
            bool ok = (idx >= 8) && (idx < H + 8);
            b1[h]  = ok ? p1[idx]  : (_Float16)0.f;
            b2[h]  = ok ? p2[idx]  : (_Float16)0.f;
            b11[h] = ok ? p11[idx] : (_Float16)0.f;
            b22[h] = ok ? p22[idx] : (_Float16)0.f;
            b12[h] = ok ? p12[idx] : (_Float16)0.f;
        }
    }
    v16h bw = make_weightB(lg, ln);

    v8f z = {0.f, 0.f, 0.f, 0.f, 0.f, 0.f, 0.f, 0.f};
    v8f mu1 = wmma_f32_16x16x32_f16(b1,  bw, z);
    v8f mu2 = wmma_f32_16x16x32_f16(b2,  bw, z);
    v8f e11 = wmma_f32_16x16x32_f16(b11, bw, z);
    v8f e22 = wmma_f32_16x16x32_f16(b22, bw, z);
    v8f e12 = wmma_f32_16x16x32_f16(b12, bw, z);

    float ss = 0.0f, cs = 0.0f;
#pragma unroll
    for (int i = 0; i < 8; ++i) {
        float m1  = mu1[i], m2 = mu2[i];
        float m1s = m1 * m1, m2s = m2 * m2, m12 = m1 * m2;
        float s1  = e11[i] - m1s;
        float s2  = e22[i] - m2s;
        float s12 = e12[i] - m12;
        float V1 = 2.0f * s12 + C2_CONST;
        float V2 = s1 + s2 + C2_CONST;
        ss += (2.0f * m12 + C1_CONST) * V1 / ((m1s + m2s + C1_CONST) * V2);
        cs += V1 / V2;
    }
#pragma unroll
    for (int off = 16; off > 0; off >>= 1) {
        ss += __shfl_down(ss, off, 32);
        cs += __shfl_down(cs, off, 32);
    }
    if (lane == 0) {
        atomicAdd(accS, ss);
        atomicAdd(accC, cs);
    }
}

// ---------------------------------------------------------------------------
// 2x2 average pool, 2 output pixels per thread (float4 in, float2 out)
// ---------------------------------------------------------------------------
__global__ void msssim_pool(const float* __restrict__ s1, const float* __restrict__ s2,
                            float* __restrict__ d1, float* __restrict__ d2,
                            int Ho, int Wo, int n2)
{
    int idx = blockIdx.x * blockDim.x + threadIdx.x;
    if (idx >= n2) return;
    int Wo2 = Wo >> 1;
    int j = idx % Wo2;
    int t = idx / Wo2;
    int y = t % Ho;
    int b = t / Ho;
    int Wi = Wo * 2;
    size_t ib = ((size_t)b * (Ho * 2) + (size_t)(2 * y)) * Wi + (size_t)(4 * j);
    vf4 a0 = *(const vf4*)(s1 + ib);
    vf4 a1 = *(const vf4*)(s1 + ib + Wi);
    vf4 b0 = *(const vf4*)(s2 + ib);
    vf4 b1 = *(const vf4*)(s2 + ib + Wi);
    vf2 oa = { 0.25f * (a0[0] + a0[1] + a1[0] + a1[1]),
               0.25f * (a0[2] + a0[3] + a1[2] + a1[3]) };
    vf2 ob = { 0.25f * (b0[0] + b0[1] + b1[0] + b1[1]),
               0.25f * (b0[2] + b0[3] + b1[2] + b1[3]) };
    size_t ob_i = ((size_t)b * Ho + (size_t)y) * Wo + (size_t)(2 * j);
    *(vf2*)(d1 + ob_i) = oa;
    *(vf2*)(d2 + ob_i) = ob;
}

// ---------------------------------------------------------------------------
// Combine 5 levels: prod(mcs[l]^w[l], l<4) * ssim[4]^w[4] / 2, clipped [0,1]
// ---------------------------------------------------------------------------
__global__ void msssim_final(const float* __restrict__ acc, float* __restrict__ out)
{
    if (threadIdx.x == 0 && blockIdx.x == 0) {
        const float w[5] = {0.0448f, 0.2856f, 0.3001f, 0.2363f, 0.1333f};
        float v = 1.0f;
        for (int l = 0; l < 5; ++l) {
            int   Hl = 512 >> l;
            float n  = 16.0f * (float)Hl * (float)Hl;
            float ms = acc[l]     / n;
            float mc = acc[5 + l] / n;
            v *= (l < 4) ? powf(mc, w[l]) : powf(ms, w[4]);
        }
        v *= 0.5f;
        out[0] = fminf(fmaxf(v, 0.0f), 1.0f);
    }
}

extern "C" void kernel_launch(void* const* d_in, const int* in_sizes, int n_in,
                              void* d_out, int out_size, void* d_ws, size_t ws_size,
                              hipStream_t stream)
{
    const float* img1 = (const float*)d_in[0];
    const float* img2 = (const float*)d_in[1];
    float* out = (float*)d_out;
    char*  ws  = (char*)d_ws;

    // ws layout: [0,256): accumulators (ssim[5], mcs[5])
    float* acc = (float*)ws;
    size_t off = 256;
    // transposed tmp channels sized for level 0: 16 imgs * W rows * (H+HPAD) halves
    const size_t NT0 = (size_t)16 * 512 * (512 + HPAD);
    _Float16* tmp[5];
    for (int c = 0; c < 5; ++c) { tmp[c] = (_Float16*)(ws + off); off += NT0 * sizeof(_Float16); }
    float* lvlA[5]; float* lvlB[5];
    lvlA[0] = nullptr; lvlB[0] = nullptr;
    for (int l = 1; l < 5; ++l) {
        size_t n = (size_t)16 * (512 >> l) * (512 >> l);
        lvlA[l] = (float*)(ws + off); off += n * sizeof(float);
        lvlB[l] = (float*)(ws + off); off += n * sizeof(float);
    }

    msssim_init<<<1, 32, 0, stream>>>(acc);

    for (int l = 0; l < 5; ++l) {
        int H = 512 >> l, W = H;
        const float* a = (l == 0) ? img1 : lvlA[l];
        const float* b = (l == 0) ? img2 : lvlB[l];
        int nTiles = 16 * (H / 16) * (W / 16);   // always a multiple of 8
        int blocks = nTiles / 8;                 // 8 waves (tiles) per 256-thread block

        msssim_hconv<<<blocks, 256, 0, stream>>>(a, b, tmp[0], tmp[1], tmp[2], tmp[3], tmp[4],
                                                 H, W, nTiles);
        msssim_vconv<<<blocks, 256, 0, stream>>>(tmp[0], tmp[1], tmp[2], tmp[3], tmp[4],
                                                 acc + l, acc + 5 + l, H, W, nTiles);
        if (l < 4) {
            int Ho = H / 2, Wo = W / 2;
            int n2 = 16 * Ho * (Wo / 2);
            msssim_pool<<<(n2 + 255) / 256, 256, 0, stream>>>(a, b, lvlA[l + 1], lvlB[l + 1],
                                                              Ho, Wo, n2);
        }
    }

    msssim_final<<<1, 1, 0, stream>>>(acc, out);
}
